// DeformConv2dPack_56315611185540
// MI455X (gfx1250) — matrix operand order repaired and checked by reading, not compiled, not run on hardware
//
#include <hip/hip_runtime.h>

typedef __attribute__((ext_vector_type(16))) __bf16 v16bf;
typedef __attribute__((ext_vector_type(8)))  float  v8f;

struct alignas(16) U4 { unsigned int a, b, c, d; };

#define CDIM  256
#define HW    4096
#define KTAPS 9
#define LDW   40   // padded LDS row stride in bf16 elements (80 bytes)

__device__ __forceinline__ int iclamp(int v, int lo, int hi) {
    return v < lo ? lo : (v > hi ? hi : v);
}

// ---------------------------------------------------------------------------
// Kernel 1: pack weight f32 (Cout, C, 3, 3) -> bf16 [k][cout][c]
// ---------------------------------------------------------------------------
__global__ void __launch_bounds__(256)
pack_weight_kernel(const float* __restrict__ weight, __bf16* __restrict__ wt)
{
    int idx = blockIdx.x * 256 + threadIdx.x;      // 9*256*256 total
    int k = idx >> 16;
    int o = (idx >> 8) & 255;
    int c = idx & 255;
    wt[idx] = (__bf16)weight[(size_t)o * (CDIM * KTAPS) + c * KTAPS + k];
}

// ---------------------------------------------------------------------------
// Kernel 2: offset (18ch) + mask (9ch, sigmoid) 3x3 conv, direct form.
// One thread per output pixel, 27 accumulators.
// ---------------------------------------------------------------------------
__global__ void __launch_bounds__(256)
offmask_kernel(const float* __restrict__ x,
               const float* __restrict__ w_off,
               const float* __restrict__ b_off,
               const float* __restrict__ w_msk,
               const float* __restrict__ b_msk,
               float* __restrict__ dyA,
               float* __restrict__ dxA,
               float* __restrict__ mA)
{
    int pix = blockIdx.x * 256 + threadIdx.x;      // 32768 total
    int b  = pix >> 12;
    int hw = pix & (HW - 1);
    int h  = hw >> 6;
    int w  = hw & 63;

    float acc[27];
#pragma unroll
    for (int j = 0; j < 18; ++j) acc[j] = b_off[j];
#pragma unroll
    for (int j = 0; j < 9; ++j)  acc[18 + j] = b_msk[j];

    const float* xb = x + (size_t)b * CDIM * HW;
    for (int c = 0; c < CDIM; ++c) {
        const float* xc = xb + (size_t)c * HW;
        float xv[9];
#pragma unroll
        for (int t = 0; t < 9; ++t) {
            int hh = h + (t / 3) - 1;
            int ww = w + (t % 3) - 1;
            bool ok = (hh >= 0) && (hh < 64) && (ww >= 0) && (ww < 64);
            xv[t] = ok ? xc[hh * 64 + ww] : 0.0f;
        }
#pragma unroll
        for (int j = 0; j < 27; ++j) {
            const float* wr = (j < 18)
                ? (w_off + (size_t)j * (CDIM * 9) + c * 9)
                : (w_msk + (size_t)(j - 18) * (CDIM * 9) + c * 9);
#pragma unroll
            for (int t = 0; t < 9; ++t) acc[j] = fmaf(xv[t], wr[t], acc[j]);
        }
    }
#pragma unroll
    for (int k = 0; k < 9; ++k) {
        int o = (b * KTAPS + k) * HW + hw;
        dyA[o] = acc[2 * k];
        dxA[o] = acc[2 * k + 1];
        mA[o]  = 1.0f / (1.0f + __expf(-acc[18 + k]));
    }
}

// ---------------------------------------------------------------------------
// Kernel 3: fused bilinear sampling + bf16 WMMA GEMM.
// Block = 256 threads (8 waves). Block computes 256(Cout) x 32(pixel) tile.
// Reduction over ck = k*256 + c in chunks of 32, k outer / c inner.
// ---------------------------------------------------------------------------
__global__ void __launch_bounds__(256)
deform_wmma_kernel(const float* __restrict__ x,
                   const float* __restrict__ dyA,
                   const float* __restrict__ dxA,
                   const float* __restrict__ mA,
                   const __bf16* __restrict__ wt,     // [9][256][256] bf16
                   const float* __restrict__ bias,
                   float* __restrict__ out)
{
    // Fragment-row-ordered LDS slabs: rows = M (or N), 32 k-values per row,
    // rows padded to 40 halves (80B) to spread LDS banks.
    __shared__ __attribute__((aligned(16))) __bf16 Wld[256 * LDW];  // 20 KB
    __shared__ __attribute__((aligned(16))) __bf16 Sld[32 * LDW];   // 2.5 KB
    __shared__ int   s_off[4][32];
    __shared__ float s_wgt[4][32];

    const int tid  = threadIdx.x;
    const int lane = tid & 31;
    const int wv   = tid >> 5;       // wave 0..7
    const int lrow = lane & 15;
    const int hsel = lane >> 4;      // 0/1: selects K-half per ISA layouts

    const int p0   = blockIdx.x * 32;     // 32 consecutive pixels, same image
    const int bImg = p0 >> 12;
    const int hw0  = p0 & (HW - 1);

    const float* xb = x + (size_t)bImg * CDIM * HW;

    v8f acc[2][2] = {};   // 256x32 tile: wave wv owns Cout [wv*32, wv*32+32)

    for (int k = 0; k < KTAPS; ++k) {
        // ---- per-tap bilinear corner offsets & (mask-folded) weights ----
        if (tid < 32) {
            int hw = hw0 + tid;
            int h  = hw >> 6;
            int w  = hw & 63;
            int o  = (bImg * KTAPS + k) * HW + hw;
            float ys = (float)h + (float)(k / 3 - 1) + dyA[o];
            float xs = (float)w + (float)(k % 3 - 1) + dxA[o];
            float fm = mA[o];
            float y0f = floorf(ys), x0f = floorf(xs);
            int iy = (int)y0f, ix = (int)x0f;
            float wy1 = ys - y0f, wx1 = xs - x0f;
            float wy0 = 1.0f - wy1, wx0 = 1.0f - wx1;
#pragma unroll
            for (int cn = 0; cn < 4; ++cn) {
                int cy = iy + (cn >> 1);
                int cx = ix + (cn & 1);
                bool ok = (cy >= 0) && (cy < 64) && (cx >= 0) && (cx < 64);
                float wgt = ((cn >> 1) ? wy1 : wy0) * ((cn & 1) ? wx1 : wx0);
                s_off[cn][tid] = iclamp(cy, 0, 63) * 64 + iclamp(cx, 0, 63);
                s_wgt[cn][tid] = ok ? wgt * fm : 0.0f;
            }
        }

        for (int cc = 0; cc < 8; ++cc) {
            __syncthreads();   // params visible; previous WMMA reads of LDS done

            // ---- stage weight chunk: Wld[cout][ci] = wt[k][cout][cc*32+ci]
            {
                const U4* src = reinterpret_cast<const U4*>(
                    wt + ((size_t)(k * 256 + tid) * 256 + cc * 32));
                U4* dst = reinterpret_cast<U4*>(Wld + tid * LDW);
#pragma unroll
                for (int i = 0; i < 4; ++i) dst[i] = src[i];
            }
            // ---- stage sampled chunk: Sld[pix][ci], 32x32 bf16
            {
                int p  = tid & 31;
                int cg = tid >> 5;                    // 4 consecutive ci each
                int o0 = s_off[0][p], o1 = s_off[1][p];
                int o2 = s_off[2][p], o3 = s_off[3][p];
                float w0 = s_wgt[0][p], w1 = s_wgt[1][p];
                float w2 = s_wgt[2][p], w3 = s_wgt[3][p];
                union { __bf16 h[4]; unsigned long long u; } pk;
#pragma unroll
                for (int u = 0; u < 4; ++u) {
                    const float* xc = xb + (size_t)(cc * 32 + cg * 4 + u) * HW;
                    float v = w0 * xc[o0] + w1 * xc[o1] + w2 * xc[o2] + w3 * xc[o3];
                    pk.h[u] = (__bf16)v;
                }
                *reinterpret_cast<unsigned long long*>(Sld + (p * LDW + cg * 4)) = pk.u;
            }
            __syncthreads();

            // ---- build fragments (two b128 LDS reads each, per ISA layouts)
            union Frag { U4 q[2]; v16bf v; };
            Frag a0, a1, b0, b1;
            {
                // A (16x32 bf16): lane holds row m=lrow; lanes 0-15 get K 0-7
                // in v0-3 & 16-23 in v4-7; lanes 16-31 get K 8-15 & 24-31.
                const char* r0 = reinterpret_cast<const char*>(Wld + (wv * 32 + lrow) * LDW);
                const char* r1 = reinterpret_cast<const char*>(Wld + (wv * 32 + 16 + lrow) * LDW);
                a0.q[0] = *reinterpret_cast<const U4*>(r0 + hsel * 16);
                a0.q[1] = *reinterpret_cast<const U4*>(r0 + 32 + hsel * 16);
                a1.q[0] = *reinterpret_cast<const U4*>(r1 + hsel * 16);
                a1.q[1] = *reinterpret_cast<const U4*>(r1 + 32 + hsel * 16);
                // B (32x16 bf16): lane holds col n=lrow; lanes 0-15 K 0-7 & 16-23,
                // lanes 16-31 K 8-15 & 24-31.
                const char* rb0 = reinterpret_cast<const char*>(Sld + lrow * LDW);
                const char* rb1 = reinterpret_cast<const char*>(Sld + (16 + lrow) * LDW);
                b0.q[0] = *reinterpret_cast<const U4*>(rb0 + hsel * 16);
                b0.q[1] = *reinterpret_cast<const U4*>(rb0 + 32 + hsel * 16);
                b1.q[0] = *reinterpret_cast<const U4*>(rb1 + hsel * 16);
                b1.q[1] = *reinterpret_cast<const U4*>(rb1 + 32 + hsel * 16);
            }
            acc[0][0] = __builtin_amdgcn_wmma_f32_16x16x32_bf16(
                false, a0.v, false, b0.v, (short)0, acc[0][0], false, false);
            acc[0][1] = __builtin_amdgcn_wmma_f32_16x16x32_bf16(
                false, a0.v, false, b1.v, (short)0, acc[0][1], false, false);
            acc[1][0] = __builtin_amdgcn_wmma_f32_16x16x32_bf16(
                false, a1.v, false, b0.v, (short)0, acc[1][0], false, false);
            acc[1][1] = __builtin_amdgcn_wmma_f32_16x16x32_bf16(
                false, a1.v, false, b1.v, (short)0, acc[1][1], false, false);
        }
    }

    // ---- write D tiles: VGPR r -> M = r + 8*hsel, N = lrow (ISA C/D layout)
    const size_t outBase = (size_t)bImg * CDIM * HW;
#pragma unroll
    for (int i = 0; i < 2; ++i) {
        int coutT = wv * 32 + i * 16;
#pragma unroll
        for (int j = 0; j < 2; ++j) {
#pragma unroll
            for (int r = 0; r < 8; ++r) {
                int cout = coutT + r + (hsel << 3);
                int hw   = hw0 + j * 16 + lrow;
                out[outBase + (size_t)cout * HW + hw] = acc[i][j][r] + bias[cout];
            }
        }
    }
}

// ---------------------------------------------------------------------------
extern "C" void kernel_launch(void* const* d_in, const int* in_sizes, int n_in,
                              void* d_out, int out_size, void* d_ws, size_t ws_size,
                              hipStream_t stream)
{
    const float* x      = (const float*)d_in[0];
    const float* w_off  = (const float*)d_in[1];
    const float* b_off  = (const float*)d_in[2];
    const float* w_msk  = (const float*)d_in[3];
    const float* b_msk  = (const float*)d_in[4];
    const float* weight = (const float*)d_in[5];
    const float* bias   = (const float*)d_in[6];
    float* out = (float*)d_out;

    const int PLANE = 8 * KTAPS * HW;            // 294912 floats
    float*  dyA = (float*)d_ws;
    float*  dxA = dyA + PLANE;
    float*  mA  = dxA + PLANE;
    __bf16* wtb = (__bf16*)(mA + PLANE);         // 589824 bf16

    pack_weight_kernel<<<(9 * 256 * 256) / 256, 256, 0, stream>>>(weight, wtb);
    offmask_kernel<<<(8 * HW) / 256, 256, 0, stream>>>(
        x, w_off, b_off, w_msk, b_msk, dyA, dxA, mA);
    deform_wmma_kernel<<<(8 * HW) / 32, 256, 0, stream>>>(
        x, dyA, dxA, mA, wtb, bias, out);
}
